// AdversarialNetwork_74225624809686
// MI455X (gfx1250) — matrix operand order repaired and checked by reading, not compiled, not run on hardware
//
#include <hip/hip_runtime.h>
#include <hip/hip_bf16.h>

typedef __attribute__((ext_vector_type(16))) _Float16 v16h;
typedef __attribute__((ext_vector_type(8)))  _Float16 v8h;
typedef __attribute__((ext_vector_type(8)))  float    v8f;
typedef __attribute__((ext_vector_type(4)))  int      v4i;

#define IMG 224
#define HW  (IMG*IMG)          // 50176
#define PS  160
#define NB  8
#define C0  32                 // conv1 input channels, zero-padded from 3

#if __has_builtin(__builtin_amdgcn_global_load_async_to_lds_b128) && \
    __has_builtin(__builtin_amdgcn_s_wait_asynccnt)
#define USE_ASYNC_LDS 1
#else
#define USE_ASYNC_LDS 0
#endif

#if USE_ASYNC_LDS
typedef __attribute__((address_space(1))) v4i glob_v4i;
typedef __attribute__((address_space(3))) v4i lds_v4i;
#endif

// Copy 16 bytes global -> LDS. Async path uses the CDNA5 ASYNCcnt-tracked
// GLOBAL_LOAD_ASYNC_TO_LDS_B128; fallback is load + ds_store.
__device__ __forceinline__ void copy16_to_lds(const _Float16* g, _Float16* l) {
#if USE_ASYNC_LDS
    __builtin_amdgcn_global_load_async_to_lds_b128(
        (glob_v4i*)(size_t)g,
        (lds_v4i*)(unsigned)(size_t)l,
        0, 0);
#else
    *(v8h*)l = *(const v8h*)g;
#endif
}

__device__ __forceinline__ void wait_async_copies() {
#if USE_ASYNC_LDS
    __builtin_amdgcn_s_wait_asynccnt(0);
#endif
}

__device__ __forceinline__ float clamp01(float v) {
    return fminf(fmaxf(v, 0.f), 1.f);
}

// ---------------------------------------------------------------------------
// 1) Resize fixed_input (200,200,3) -> patch (3,160,160), linear w/ antialias
// ---------------------------------------------------------------------------
__global__ void resize_kernel(const float* __restrict__ fixed, float* __restrict__ patch) {
    int idx = blockIdx.x * blockDim.x + threadIdx.x;
    if (idx >= 3 * PS * PS) return;
    int xo = idx % PS;
    int t  = idx / PS;
    int yo = t % PS;
    int c  = t / PS;
    const float scale = 160.f / 200.f;
    const float inv   = 200.f / 160.f;
    float sy = (yo + 0.5f) * inv - 0.5f;
    float sx = (xo + 0.5f) * inv - 0.5f;
    int y0 = (int)floorf(sy) - 1;
    int x0 = (int)floorf(sx) - 1;
    float acc = 0.f, wsum = 0.f;
    #pragma unroll
    for (int dy = 0; dy < 4; ++dy) {
        int iy = y0 + dy;
        if (iy < 0 || iy >= 200) continue;
        float wy = 1.f - fabsf((float)iy - sy) * scale;
        if (wy <= 0.f) continue;
        #pragma unroll
        for (int dx = 0; dx < 4; ++dx) {
            int ix = x0 + dx;
            if (ix < 0 || ix >= 200) continue;
            float wx = 1.f - fabsf((float)ix - sx) * scale;
            if (wx <= 0.f) continue;
            float w = wy * wx;
            acc  += w * fixed[(ix * 200 + iy) * 3 + c];   // swapaxes(0,2)
            wsum += w;
        }
    }
    patch[idx] = acc / wsum;
}

// ---------------------------------------------------------------------------
// 2) Composite rotated patch under circular mask + brightness -> fp32 NCHW
// ---------------------------------------------------------------------------
__global__ void composite_kernel(const float* __restrict__ x,
                                 const float* __restrict__ patch,
                                 float* __restrict__ img) {
    int idx = blockIdx.x * blockDim.x + threadIdx.x;
    if (idx >= NB * 3 * HW) return;
    int xw = idx % IMG;
    int t  = idx / IMG;
    int yw = t % IMG;
    t /= IMG;
    int c  = t % 3;
    float v = x[idx];
    int py = yw - 30;   // V_SPLIT
    int px = xw - 40;   // H_SPLIT
    if (py >= 0 && py < PS && px >= 0 && px < PS) {
        float my = (float)py - 80.f, mx = (float)px - 80.f;
        if (my * my + mx * mx <= 6400.f) {
            const float ct = 0.9396926207859084f;
            const float st = 0.3420201433256687f;
            float dx = (float)px - 79.5f, dy = (float)py - 79.5f;
            float sx =  ct * dx + st * dy + 79.5f;
            float sy = -st * dx + ct * dy + 79.5f;
            int ix = (int)rintf(sx);
            int iy = (int)rintf(sy);
            float pv = 0.f;
            if (ix >= 0 && ix < PS && iy >= 0 && iy < PS)
                pv = patch[(c * PS + iy) * PS + ix];
            v = pv;
        }
    }
    img[idx] = clamp01(v * 1.1f);
}

// ---------------------------------------------------------------------------
// 3) Per-image gray mean
// ---------------------------------------------------------------------------
__global__ void mean_kernel(const float* __restrict__ img, float* __restrict__ gmean) {
    __shared__ float sdata[256];
    int n = blockIdx.x;
    const float* base = img + (size_t)n * 3 * HW;
    float s = 0.f;
    for (int i = threadIdx.x; i < HW; i += 256) {
        float r = base[i], g = base[HW + i], b = base[2 * HW + i];
        s += 0.2989f * r + 0.587f * g + 0.114f * b;
    }
    sdata[threadIdx.x] = s;
    __syncthreads();
    for (int st = 128; st > 0; st >>= 1) {
        if (threadIdx.x < st) sdata[threadIdx.x] += sdata[threadIdx.x + st];
        __syncthreads();
    }
    if (threadIdx.x == 0) gmean[n] = sdata[0] / (float)HW;
}

// ---------------------------------------------------------------------------
// 4) Contrast + saturation + hue; emit fp16 NHWC [8][224][224][32] (pad zeros)
// ---------------------------------------------------------------------------
__global__ void color_kernel(const float* __restrict__ img,
                             const float* __restrict__ gmean,
                             _Float16* __restrict__ act) {
    int idx = blockIdx.x * blockDim.x + threadIdx.x;
    if (idx >= NB * HW) return;
    int pix = idx % HW;
    int n   = idx / HW;
    const float* base = img + (size_t)n * 3 * HW;
    float r = base[pix], g = base[HW + pix], b = base[2 * HW + pix];
    float m = gmean[n];
    r = clamp01(1.3f * r - 0.3f * m);
    g = clamp01(1.3f * g - 0.3f * m);
    b = clamp01(1.3f * b - 0.3f * m);
    float gray = 0.2989f * r + 0.587f * g + 0.114f * b;
    r = clamp01(0.9f * r + 0.1f * gray);
    g = clamp01(0.9f * g + 0.1f * gray);
    b = clamp01(0.9f * b + 0.1f * gray);
    float maxc = fmaxf(r, fmaxf(g, b));
    float minc = fminf(r, fminf(g, b));
    bool  eqc  = (maxc == minc);
    float cr   = maxc - minc;
    float s    = cr / (eqc ? 1.f : maxc);
    float crd  = eqc ? 1.f : cr;
    float rc = (maxc - r) / crd, gc = (maxc - g) / crd, bc = (maxc - b) / crd;
    float h;
    if (maxc == r)      h = bc - gc;
    else if (maxc == g) h = 2.f + rc - bc;
    else                h = 4.f + gc - rc;
    h = fmodf(h / 6.f + 1.f, 1.f);
    h = fmodf(h + 0.05f, 1.f);
    float f6 = h * 6.f;
    float fi = floorf(f6);
    float f  = f6 - fi;
    int   i  = ((int)fi) % 6;
    float v  = maxc;
    float p  = v * (1.f - s);
    float q  = v * (1.f - s * f);
    float tt = v * (1.f - s * (1.f - f));
    float rr, gg, bb;
    switch (i) {
        case 0: rr = v;  gg = tt; bb = p;  break;
        case 1: rr = q;  gg = v;  bb = p;  break;
        case 2: rr = p;  gg = v;  bb = tt; break;
        case 3: rr = p;  gg = q;  bb = v;  break;
        case 4: rr = tt; gg = p;  bb = v;  break;
        default: rr = v; gg = p;  bb = q;  break;
    }
    _Float16* abase = act + (size_t)idx * C0;
    v8h z = {};
    *(v8h*)(abase)      = z;
    *(v8h*)(abase + 8)  = z;
    *(v8h*)(abase + 16) = z;
    *(v8h*)(abase + 24) = z;
    abase[0] = (_Float16)rr;
    abase[1] = (_Float16)gg;
    abase[2] = (_Float16)bb;
}

// ---------------------------------------------------------------------------
// 5) Pack conv weights fp32 OIHW -> fp16 A-fragment order
// ---------------------------------------------------------------------------
__global__ void pack_conv_w(const float* __restrict__ W, _Float16* __restrict__ P,
                            int Cin, int Cinp, int Cout) {
    int KT = Cinp >> 5;
    int total = (Cout >> 4) * 9 * KT * 512;
    int idx = blockIdx.x * blockDim.x + threadIdx.x;
    if (idx >= total) return;
    int e    = idx & 15;
    int t    = idx >> 4;
    int lane = t & 31;
    t >>= 5;
    int kt  = t % KT; t /= KT;
    int tap = t % 9;
    int ot  = t / 9;
    int r = tap / 3, c = tap - r * 3;
    int och = ot * 16 + (lane & 15);
    int k0  = (lane >> 4) * 16;
    int ci  = kt * 32 + k0 + e;
    float v = 0.f;
    if (ci < Cin) v = W[(((size_t)och * Cin + ci) * 3 + r) * 3 + c];
    P[idx] = (_Float16)v;
}

// ---------------------------------------------------------------------------
// 6) Conv 3x3 SAME, NHWC, implicit GEMM via v_wmma_f32_16x16x32_f16.
//    Wave: 16 och x 64 pixels (4 N-tiles share one A fragment).
//    A tile for (tap) cooperatively staged in LDS via async global->LDS
//    (shared by all 8 waves of the block), then read with ds_load.
// ---------------------------------------------------------------------------
#define PIXT 4
__global__ void __launch_bounds__(256) conv_wmma_kernel(
    const _Float16* __restrict__ in, _Float16* __restrict__ out,
    const _Float16* __restrict__ P, const float* __restrict__ bias,
    int Cin, int H, int Wd, int Cout)
{
    __shared__ __align__(16) _Float16 shA[16 * 512];   // up to KT=16 -> 16KB

    const int KT   = Cin >> 5;
    const int lane = threadIdx.x & 31;
    const int wave = threadIdx.x >> 5;
    const int hi   = lane >> 4;
    const int col  = lane & 15;

    const int ot       = blockIdx.x;                 // och tile (shared by block)
    const int totalPix = NB * H * Wd;
    const int pixBase  = (blockIdx.y * 8 + wave) * (16 * PIXT);
    const bool waveActive = (pixBase < totalPix);    // wave-uniform

    // decode per-tile / per-lane pixel coords (clamped: safe for idle lanes)
    int nA[PIXT], yA[PIXT], xA[PIXT];
    bool tval[PIXT];
    const int hw = H * Wd;
    #pragma unroll
    for (int tl = 0; tl < PIXT; ++tl) {
        int pb = pixBase + tl * 16;
        tval[tl] = waveActive && (pb < totalPix);    // wave-uniform
        int pix = pb + col;
        if (pix >= totalPix) pix = totalPix - 1;
        int n   = pix / hw;
        int rem = pix - n * hw;
        int y   = rem / Wd;
        nA[tl] = n; yA[tl] = y; xA[tl] = rem - y * Wd;
    }

    v8f acc[PIXT];
    #pragma unroll
    for (int tl = 0; tl < PIXT; ++tl) acc[tl] = (v8f){};

    const _Float16* Pt = P + (size_t)ot * 9 * KT * 512;
    const int kb0    = hi * 16;
    const int chunks = KT * 64;                      // 16B chunks per tap tile

    for (int tap = 0; tap < 9; ++tap) {
        __syncthreads();                             // LDS reuse fence
        for (int chk = threadIdx.x; chk < chunks; chk += 256)
            copy16_to_lds(Pt + ((size_t)tap * KT * 512) + chk * 8, shA + chk * 8);
        wait_async_copies();
        __syncthreads();

        if (waveActive) {
            int r = tap / 3, c = tap - r * 3;
            for (int kt = 0; kt < KT; ++kt) {
                v16h a = *(const v16h*)(shA + kt * 512 + lane * 16);  // ds_load
                #pragma unroll
                for (int tl = 0; tl < PIXT; ++tl) {
                    if (!tval[tl]) continue;         // wave-uniform
                    int iy = yA[tl] + r - 1;
                    int ix = xA[tl] + c - 1;
                    v16h b = {};
                    if (iy >= 0 && iy < H && ix >= 0 && ix < Wd) {
                        const _Float16* bp = in +
                            ((((size_t)nA[tl] * H + iy) * Wd + ix) * Cin + kt * 32 + kb0);
                        b = *(const v16h*)bp;
                    }
                    acc[tl] = __builtin_amdgcn_wmma_f32_16x16x32_f16(
                        false, a, false, b, (short)0, acc[tl], false, false);
                }
            }
        }
    }

    if (!waveActive) return;
    const int ocb = ot * 16 + hi * 8;
    float bb[8];
    #pragma unroll
    for (int j = 0; j < 8; ++j) bb[j] = bias[ocb + j];
    #pragma unroll
    for (int tl = 0; tl < PIXT; ++tl) {
        if (!tval[tl]) continue;
        v8h o;
        #pragma unroll
        for (int j = 0; j < 8; ++j) {
            float v = acc[tl][j] + bb[j];
            o[j] = (_Float16)(v > 0.f ? v : 0.f);
        }
        _Float16* op = out +
            ((((size_t)nA[tl] * H + yA[tl]) * Wd + xA[tl]) * Cout + ocb);
        *(v8h*)op = o;
    }
}

// ---------------------------------------------------------------------------
// 7) 2x2 max-pool, NHWC fp16
// ---------------------------------------------------------------------------
__global__ void pool_kernel(const _Float16* __restrict__ in, _Float16* __restrict__ out,
                            int C, int Hin, int Win) {
    int Ho = Hin >> 1, Wo = Win >> 1;
    int total = NB * C * Ho * Wo;
    int idx = blockIdx.x * blockDim.x + threadIdx.x;
    if (idx >= total) return;
    int c  = idx % C;
    int t  = idx / C;
    int xo = t % Wo;
    t /= Wo;
    int yo = t % Ho;
    int n  = t / Ho;
    const _Float16* p = in + ((((size_t)n * Hin + yo * 2) * Win + xo * 2) * C + c);
    size_t rs = (size_t)Win * C;
    float a = (float)p[0], b = (float)p[C], d = (float)p[rs], e = (float)p[rs + C];
    out[idx] = (_Float16)fmaxf(fmaxf(a, b), fmaxf(d, e));
}

// ---------------------------------------------------------------------------
// 8) Repack pool5 NHWC [8][7][7][512] -> NCHW-flat [8][25088]
// ---------------------------------------------------------------------------
__global__ void fc_repack_kernel(const _Float16* __restrict__ in, _Float16* __restrict__ out) {
    int idx = blockIdx.x * blockDim.x + threadIdx.x;
    if (idx >= NB * 25088) return;
    int k = idx % 25088;
    int n = idx / 25088;
    int c   = k / 49;
    int rem = k - c * 49;
    out[idx] = in[((size_t)n * 49 + rem) * 512 + c];
}

// ---------------------------------------------------------------------------
// 9) FC layer as WMMA GEMM: M=16 (batch 8 zero-padded), N=16 features / wave
// ---------------------------------------------------------------------------
__global__ void __launch_bounds__(128) fc_wmma_kernel(
    const _Float16* __restrict__ in, const float* __restrict__ Wt,
    const float* __restrict__ bias, _Float16* __restrict__ outH,
    float* __restrict__ outF, int K, int F, int doRelu)
{
    const int lane = threadIdx.x & 31;
    const int wave = threadIdx.x >> 5;
    const int hi   = lane >> 4;
    const int col  = lane & 15;
    const int Fpad = (F + 15) & ~15;
    const int fBase = (blockIdx.x * 4 + wave) * 16;
    if (fBase >= Fpad) return;        // wave-uniform

    const int f = fBase + col;
    const float* wrow = (f < F) ? (Wt + (size_t)f * K) : Wt;
    const _Float16* arow = in + (size_t)col * K;

    v8f acc = {};
    for (int kk = 0; kk < K; kk += 32) {
        v16h a, b;
        int k0 = kk + hi * 8;
        #pragma unroll
        for (int i = 0; i < 8; ++i) {
            a[i]     = (col < NB) ? arow[k0 + i]      : (_Float16)0.f;
            a[8 + i] = (col < NB) ? arow[k0 + 16 + i] : (_Float16)0.f;
        }
        int kb0 = kk + hi * 16;
        #pragma unroll
        for (int i = 0; i < 16; ++i) {
            b[i] = (f < F) ? (_Float16)wrow[kk + hi * 8 + (i < 8 ? i : i + 8)] : (_Float16)0.f;
        }
        acc = __builtin_amdgcn_wmma_f32_16x16x32_f16(
            false, a, false, b, (short)0, acc, false, false);
    }

    #pragma unroll
    for (int j = 0; j < 8; ++j) {
        int row = j + hi * 8;
        if (row < NB && f < F) {
            float v = acc[j] + bias[f];
            if (doRelu) v = v > 0.f ? v : 0.f;
            if (outF) outF[(size_t)row * F + f] = v;
            else      outH[(size_t)row * F + f] = (_Float16)v;
        }
    }
}

// ---------------------------------------------------------------------------
// Host orchestration
// ---------------------------------------------------------------------------
extern "C" void kernel_launch(void* const* d_in, const int* in_sizes, int n_in,
                              void* d_out, int out_size, void* d_ws, size_t ws_size,
                              hipStream_t stream) {
    (void)in_sizes; (void)n_in; (void)out_size; (void)ws_size;
    const float* x_in  = (const float*)d_in[0];
    const float* fixed = (const float*)d_in[1];

    char* ws = (char*)d_ws;
    size_t off = 0;
    auto alloc = [&](size_t bytes) -> void* {
        void* p = ws + off;
        off += (bytes + 255) & ~(size_t)255;
        return p;
    };
    float*    patch  = (float*)alloc((size_t)3 * PS * PS * 4);
    float*    imgbuf = (float*)alloc((size_t)NB * 3 * HW * 4);
    float*    gmean  = (float*)alloc(NB * 4);
    _Float16* actA   = (_Float16*)alloc((size_t)NB * 64 * HW * 2);  // 51.4 MB
    _Float16* actB   = (_Float16*)alloc((size_t)NB * 64 * HW * 2);  // 51.4 MB
    _Float16* fcin   = (_Float16*)alloc((size_t)NB * 25088 * 2);
    _Float16* fcb1   = (_Float16*)alloc((size_t)NB * 4096 * 2);
    _Float16* fcb2   = (_Float16*)alloc((size_t)NB * 4096 * 2);

    static const int cfg[18] = {64, 64, 0, 128, 128, 0, 256, 256, 256, 0,
                                512, 512, 512, 0, 512, 512, 512, 0};

    // ---- pack all conv weights ----
    _Float16* packed[13];
    {
        int Cin = 3, Cinp = C0, pi = 0;
        for (int i = 0; i < 18; ++i) {
            if (cfg[i] == 0) continue;
            int Cout = cfg[i];
            int total = (Cout / 16) * 9 * (Cinp / 32) * 512;
            packed[pi] = (_Float16*)alloc((size_t)total * 2);
            pack_conv_w<<<(total + 255) / 256, 256, 0, stream>>>(
                (const float*)d_in[2 + 2 * pi], packed[pi], Cin, Cinp, Cout);
            ++pi;
            Cin = Cout; Cinp = Cout;
        }
    }

    // ---- preprocessing ----
    resize_kernel<<<(3 * PS * PS + 255) / 256, 256, 0, stream>>>(fixed, patch);
    composite_kernel<<<(NB * 3 * HW + 255) / 256, 256, 0, stream>>>(x_in, patch, imgbuf);
    mean_kernel<<<NB, 256, 0, stream>>>(imgbuf, gmean);
    color_kernel<<<(NB * HW + 255) / 256, 256, 0, stream>>>(imgbuf, gmean, actA);

    // ---- VGG16 convs (NHWC fp16) ----
    int H = IMG, Wd = IMG, Cin = C0, pi = 0;
    _Float16* cur = actA;
    _Float16* nxt = actB;
    for (int i = 0; i < 18; ++i) {
        if (cfg[i] == 0) {
            int Ho = H / 2, Wo = Wd / 2;
            int total = NB * Cin * Ho * Wo;
            pool_kernel<<<(total + 255) / 256, 256, 0, stream>>>(cur, nxt, Cin, H, Wd);
            _Float16* t = cur; cur = nxt; nxt = t;
            H = Ho; Wd = Wo;
        } else {
            int Cout = cfg[i];
            const float* bp = (const float*)d_in[3 + 2 * pi];
            int pixPerBlock = 8 * 16 * PIXT;   // 8 waves x 64 pixels
            dim3 grid(Cout / 16, (NB * H * Wd + pixPerBlock - 1) / pixPerBlock);
            conv_wmma_kernel<<<grid, 256, 0, stream>>>(cur, nxt, packed[pi], bp,
                                                       Cin, H, Wd, Cout);
            ++pi;
            _Float16* t = cur; cur = nxt; nxt = t;
            Cin = Cout;
        }
    }
    // cur = pool5 NHWC [8][7][7][512]

    // ---- FC layers ----
    fc_repack_kernel<<<(NB * 25088 + 255) / 256, 256, 0, stream>>>(cur, fcin);
    const float* W1 = (const float*)d_in[28]; const float* B1 = (const float*)d_in[29];
    const float* W2 = (const float*)d_in[30]; const float* B2 = (const float*)d_in[31];
    const float* W3 = (const float*)d_in[32]; const float* B3 = (const float*)d_in[33];

    fc_wmma_kernel<<<64, 128, 0, stream>>>(fcin, W1, B1, fcb1, (float*)nullptr, 25088, 4096, 1);
    fc_wmma_kernel<<<64, 128, 0, stream>>>(fcb1, W2, B2, fcb2, (float*)nullptr, 4096, 4096, 1);
    fc_wmma_kernel<<<16, 128, 0, stream>>>(fcb2, W3, B3, (_Float16*)nullptr, (float*)d_out, 4096, 1000, 0);
}
